// AttnAdaIN_5669356832561
// MI455X (gfx1250) — matrix operand order repaired and checked by reading, not compiled, not run on hardware
//
#include <hip/hip_runtime.h>
#include <stdint.h>

// Problem constants (from reference: B=4, C=512, H=W=64)
#define BATCH 4
#define CH    512
#define HWSZ  4096
#define EPSV  1e-5f

typedef __attribute__((ext_vector_type(16))) __bf16   v16bf;
typedef __attribute__((ext_vector_type(8)))  float    v8f;
typedef __attribute__((ext_vector_type(4)))  uint32_t v4u;
typedef __attribute__((ext_vector_type(4)))  unsigned int u32x4;
typedef __attribute__((ext_vector_type(8)))  int      i32x8;
typedef __attribute__((ext_vector_type(4)))  int      i32x4;

#if __has_builtin(__builtin_amdgcn_tensor_load_to_lds) && \
    __has_builtin(__builtin_amdgcn_s_wait_tensorcnt)
#define HAVE_TDM 1
#endif

union FragBF {
  v16bf bf;
  v4u   q[2];
  unsigned short s[16];
};

__device__ inline unsigned short f32_to_bf16(float f) {
  uint32_t u = __float_as_uint(f);
  uint32_t r = u + 0x7FFFu + ((u >> 16) & 1u);   // round-to-nearest-even
  return (unsigned short)(r >> 16);
}
__device__ inline float bf16_to_f32(unsigned short s) {
  return __uint_as_float(((uint32_t)s) << 16);
}
__device__ inline v4u ld128g(const unsigned short* p) {
  return *reinterpret_cast<const v4u*>(p);
}

// ---------------------------------------------------------------------------
// Kernel 0: convert the three 512x512 weight matrices to bf16
// ---------------------------------------------------------------------------
__global__ void wconv_kernel(const float* __restrict__ fw, const float* __restrict__ gw,
                             const float* __restrict__ hw,
                             unsigned short* __restrict__ wf, unsigned short* __restrict__ wg,
                             unsigned short* __restrict__ wh) {
  int i = blockIdx.x * blockDim.x + threadIdx.x;
  if (i < CH * CH) {
    wf[i] = f32_to_bf16(fw[i]);
    wg[i] = f32_to_bf16(gw[i]);
    wh[i] = f32_to_bf16(hw[i]);
  }
}

// ---------------------------------------------------------------------------
// Kernel 1: per-(b,c) content mean and 1/std (unbiased var, +EPS under sqrt)
// ---------------------------------------------------------------------------
__global__ __launch_bounds__(256)
void stats_kernel(const float* __restrict__ content, float* __restrict__ cmean,
                  float* __restrict__ cistd) {
  int bc = blockIdx.x;                 // b*CH + c
  const float* p = content + (size_t)bc * HWSZ;
  float s = 0.f, ss = 0.f;
  for (int i = threadIdx.x; i < HWSZ; i += 256) { float v = p[i]; s += v; ss += v * v; }
  __shared__ float rs[256], rss[256];
  rs[threadIdx.x] = s; rss[threadIdx.x] = ss;
  __syncthreads();
  for (int off = 128; off > 0; off >>= 1) {
    if (threadIdx.x < off) { rs[threadIdx.x] += rs[threadIdx.x + off];
                             rss[threadIdx.x] += rss[threadIdx.x + off]; }
    __syncthreads();
  }
  if (threadIdx.x == 0) {
    float m   = rs[0] / (float)HWSZ;
    float var = (rss[0] - (float)HWSZ * m * m) / (float)(HWSZ - 1);
    cmean[bc] = m;
    cistd[bc] = rsqrtf(var + EPSV);
  }
}

// ---------------------------------------------------------------------------
// Kernel 2: 1x1-conv projection GEMM on WMMA bf16.
//   out[o,p] = sum_c w[o,c]*x[c,p] + bias[o]
//   mode 0/1: store query-major  out0[b][p][o]           (Q and K^T)
//   mode 2  : store channel-major out0[b][o][p] = V, out1 = V*V
// ---------------------------------------------------------------------------
__global__ __launch_bounds__(256)
void proj_kernel(const float* __restrict__ x,             // [B][CH][HWSZ] f32
                 const unsigned short* __restrict__ wbf,  // [CH][CH] bf16
                 const float* __restrict__ bias,          // [CH]
                 unsigned short* __restrict__ out0,
                 unsigned short* __restrict__ out1,
                 int mode) {
  __shared__ __align__(16) unsigned short Bs[8][16][40];  // [wave][p][c] bf16, padded row
  const int tid  = threadIdx.x;
  const int wv   = tid >> 5;
  const int lane = tid & 31;
  const int lo   = lane & 15;
  const int hi   = lane >> 4;
  const int p0   = blockIdx.x * 128 + wv * 16;
  const int o0   = blockIdx.y * 16;
  const int b    = blockIdx.z;
  const float* xb = x + (size_t)b * CH * HWSZ;

  v8f acc = {0.f,0.f,0.f,0.f,0.f,0.f,0.f,0.f};

  for (int cs = 0; cs < CH; cs += 32) {
    // stage B tile (32c x 16p), transposed into [p][c] bf16 in LDS
    {
      const float* src = xb + (size_t)(cs + lane) * HWSZ + p0;  // one c-row per lane
      float vals[16];
      #pragma unroll
      for (int j = 0; j < 16; ++j) vals[j] = src[j];
      #pragma unroll
      for (int j = 0; j < 16; ++j) Bs[wv][j][lane] = f32_to_bf16(vals[j]);
    }
    __syncthreads();

    FragBF a, bb;
    // A fragment (16o x 32c) from global bf16 weights, row-major in c
    const unsigned short* wrow = wbf + (size_t)(o0 + lo) * CH + cs + hi * 16;
    a.q[0] = ld128g(wrow);
    a.q[1] = ld128g(wrow + 8);
    // B fragment (32c x 16p) from LDS transpose tile
    const unsigned short* brow = &Bs[wv][lo][hi * 16];
    bb.q[0] = *reinterpret_cast<const v4u*>(brow);
    bb.q[1] = *reinterpret_cast<const v4u*>(brow + 8);

    acc = __builtin_amdgcn_wmma_f32_16x16x32_bf16(false, a.bf, false, bb.bf,
                                                  (short)0, acc, false, false);
    __syncthreads();
  }

  if (mode < 2) {
    // D tile: lane -> p = p0+lo, VGPR r -> o = o0 + hi*8 + r  (8 consecutive o)
    union { unsigned short s[8]; v4u q; } packed;
    #pragma unroll
    for (int r = 0; r < 8; ++r)
      packed.s[r] = f32_to_bf16(acc[r] + bias[o0 + hi * 8 + r]);
    unsigned short* dst = out0 + ((size_t)b * HWSZ + (p0 + lo)) * CH + o0 + hi * 8;
    *reinterpret_cast<v4u*>(dst) = packed.q;
  } else {
    #pragma unroll
    for (int r = 0; r < 8; ++r) {
      int o = o0 + hi * 8 + r;
      float v = acc[r] + bias[o];
      size_t idx = ((size_t)b * CH + o) * HWSZ + p0 + lo;
      out0[idx] = f32_to_bf16(v);
      out1[idx] = f32_to_bf16(v * v);
    }
  }
}

// ---------------------------------------------------------------------------
// Kernel 3: fused attention + AdaIN epilogue. One block (8 waves) / 16 queries.
// Q tile staged to LDS by the Tensor Data Mover (row pad done by TDM pad
// logic); full 16x4096 logit row lives in LDS bf16 (320KB LDS). Exact softmax,
// then mean = P V, second = P V^2, std = sqrt(relu(second-mean^2)), fused
// AdaIN write.
// ---------------------------------------------------------------------------
#define QS_STRIDE 520    // 512 data + 8 pad bf16 -> 1040B rows (16B aligned)
#define PS_STRIDE 4104   // 4096 data + 8 pad bf16 -> 8208B rows (16B aligned)
#define ATTN_LDS_BYTES ((16 * QS_STRIDE + 16 * PS_STRIDE) * 2)

__global__ __launch_bounds__(256)
void attn_kernel(const unsigned short* __restrict__ Q,    // [B][HWSZ][CH] bf16
                 const unsigned short* __restrict__ Kt,   // [B][HWSZ][CH] bf16 (= G^T)
                 const unsigned short* __restrict__ V,    // [B][CH][HWSZ] bf16
                 const unsigned short* __restrict__ V2,   // [B][CH][HWSZ] bf16
                 const float* __restrict__ content,
                 const float* __restrict__ cmean,
                 const float* __restrict__ cistd,
                 float* __restrict__ out) {
  extern __shared__ __align__(16) unsigned short smem[];
  unsigned short* Qs = smem;                    // 16 x QS_STRIDE
  unsigned short* Ps = smem + 16 * QS_STRIDE;   // 16 x PS_STRIDE

  const int tid  = threadIdx.x;
  const int wv   = tid >> 5;
  const int lane = tid & 31;
  const int lo   = lane & 15;
  const int hi   = lane >> 4;
  const int q0   = blockIdx.x * 16;
  const int b    = blockIdx.y;

  const unsigned short* Qb  = Q  + ((size_t)b * HWSZ + q0) * CH;
  const unsigned short* Ktb = Kt + (size_t)b * HWSZ * CH;
  const unsigned short* Vb  = V  + (size_t)b * CH * HWSZ;
  const unsigned short* V2b = V2 + (size_t)b * CH * HWSZ;

  // ---- stage the 16 query rows (16 x 512 bf16, contiguous in global) into
  //      LDS with 8-element row pad.
#ifdef HAVE_TDM
  if (wv == 0) {
    // Tensor DMA descriptor (D#): 2D tile 512(x) x 16(y), 2-byte elements,
    // rows contiguous (stride 512), LDS pad: 4 DWORDs every 256 DWORDs
    // (pad_interval=7 -> 256 dw = 512 bf16, pad_amount=3 -> 4 dw = 8 bf16),
    // which yields exactly the 520-element LDS row stride.
    unsigned lds_off = (unsigned)(size_t)(void*)Qs;           // addr[31:0] = LDS offset
    unsigned long long ga = (unsigned long long)(size_t)(const void*)Qb;
    u32x4 g0;
    g0[0] = 1u;                                               // count=1, user mode
    g0[1] = lds_off;                                          // lds_addr
    g0[2] = (unsigned)(ga & 0xFFFFFFFFu);                     // global_addr[31:0]
    g0[3] = (unsigned)((ga >> 32) & 0x01FFFFFFu) | (2u << 30);// addr[56:32] | type=2
    i32x8 g1;
    g1[0] = (int)((1u << 16) |        // data_size = 1 -> 2 bytes
                  (1u << 20) |        // pad_enable
                  (7u << 22) |        // pad_interval: 256 DWORDs
                  (3u << 25));        // pad_amount: 4 DWORDs
    g1[1] = (int)((512u & 0xFFFFu) << 16);   // tensor_dim0[15:0] at bits 63:48
    g1[2] = (int)(16u << 16);                // tensor_dim0[31:16]=0 | tensor_dim1[15:0]=16
    g1[3] = (int)(512u << 16);               // tensor_dim1[31:16]=0 | tile_dim0=512
    g1[4] = (int)(16u);                      // tile_dim1=16 | tile_dim2=0
    g1[5] = (int)(512u);                     // tensor_dim0_stride[31:0]=512
    g1[6] = 0;                               // stride hi | tensor_dim1_stride lo
    g1[7] = 0;
    i32x4 gz4 = {0, 0, 0, 0};                // D# groups 2/3 (unused dims, zero)
    i32x8 gz8 = {0, 0, 0, 0, 0, 0, 0, 0};    // extra group in 6-arg builtin form
    __builtin_amdgcn_tensor_load_to_lds(g0, g1, gz4, gz4, gz8, 0);
    __builtin_amdgcn_s_wait_tensorcnt(0);    // TENSORcnt == 0 before LDS reads
  }
#else
  for (int i = tid; i < 16 * 64; i += 256) {
    int row = i >> 6, chunk = i & 63;
    *reinterpret_cast<v4u*>(&Qs[row * QS_STRIDE + chunk * 8]) =
        ld128g(Qb + (size_t)row * CH + chunk * 8);
  }
#endif
  __syncthreads();

  // ---- phase 1: logits S[16 x 4096] -> LDS bf16. Wave wv owns k in [wv*512, wv*512+512)
  for (int ktl = 0; ktl < 32; ++ktl) {
    int k0 = wv * 512 + ktl * 16;
    v8f s = {0.f,0.f,0.f,0.f,0.f,0.f,0.f,0.f};
    for (int cs = 0; cs < CH; cs += 32) {
      FragBF a, bb;
      const unsigned short* arow = &Qs[lo * QS_STRIDE + cs + hi * 16];
      a.q[0] = *reinterpret_cast<const v4u*>(arow);
      a.q[1] = *reinterpret_cast<const v4u*>(arow + 8);
      const unsigned short* brow = Ktb + (size_t)(k0 + lo) * CH + cs + hi * 16;
      bb.q[0] = ld128g(brow);
      bb.q[1] = ld128g(brow + 8);
      s = __builtin_amdgcn_wmma_f32_16x16x32_bf16(false, a.bf, false, bb.bf,
                                                  (short)0, s, false, false);
    }
    #pragma unroll
    for (int r = 0; r < 8; ++r) {
      int q = hi * 8 + r;
      Ps[q * PS_STRIDE + k0 + lo] = f32_to_bf16(s[r]);
    }
  }
  __syncthreads();

  // ---- phase 2: exact softmax per row; wave wv handles rows 2*wv, 2*wv+1
  for (int rr = 0; rr < 2; ++rr) {
    int q = wv * 2 + rr;
    unsigned short* row = &Ps[q * PS_STRIDE];
    float m = -3.0e38f;
    for (int i = lane; i < HWSZ; i += 32) m = fmaxf(m, bf16_to_f32(row[i]));
    #pragma unroll
    for (int off = 16; off > 0; off >>= 1) m = fmaxf(m, __shfl_xor(m, off, 32));
    float sum = 0.f;
    for (int i = lane; i < HWSZ; i += 32) sum += __expf(bf16_to_f32(row[i]) - m);
    #pragma unroll
    for (int off = 16; off > 0; off >>= 1) sum += __shfl_xor(sum, off, 32);
    float inv = 1.0f / sum;
    for (int i = lane; i < HWSZ; i += 32)
      row[i] = f32_to_bf16(__expf(bf16_to_f32(row[i]) - m) * inv);
  }
  __syncthreads();

  // ---- phase 3: mean = P V, second = P V^2. Wave wv owns channels [wv*64, wv*64+64)
  v8f z = {0.f,0.f,0.f,0.f,0.f,0.f,0.f,0.f};
  v8f accM[4] = {z, z, z, z};
  v8f accS[4] = {z, z, z, z};
  const int cbase = wv * 64;
  for (int k0 = 0; k0 < HWSZ; k0 += 32) {
    FragBF a;
    const unsigned short* arow = &Ps[lo * PS_STRIDE + k0 + hi * 16];
    a.q[0] = *reinterpret_cast<const v4u*>(arow);
    a.q[1] = *reinterpret_cast<const v4u*>(arow + 8);
    #pragma unroll
    for (int t = 0; t < 4; ++t) {
      int c = cbase + t * 16 + lo;
      const unsigned short* vrow  = Vb  + (size_t)c * HWSZ + k0 + hi * 16;
      const unsigned short* v2row = V2b + (size_t)c * HWSZ + k0 + hi * 16;
      __builtin_prefetch(vrow + 32, 0, 0);     // global_prefetch_b8, next k-step
      __builtin_prefetch(v2row + 32, 0, 0);
      FragBF bv, bw;
      bv.q[0] = ld128g(vrow);   bv.q[1] = ld128g(vrow + 8);
      bw.q[0] = ld128g(v2row);  bw.q[1] = ld128g(v2row + 8);
      accM[t] = __builtin_amdgcn_wmma_f32_16x16x32_bf16(false, a.bf, false, bv.bf,
                                                        (short)0, accM[t], false, false);
      accS[t] = __builtin_amdgcn_wmma_f32_16x16x32_bf16(false, a.bf, false, bw.bf,
                                                        (short)0, accS[t], false, false);
    }
  }

  // ---- phase 4: AdaIN epilogue, D tile: lane -> c = cbase+t*16+lo, r -> q = q0+hi*8+r
  #pragma unroll
  for (int t = 0; t < 4; ++t) {
    int c = cbase + t * 16 + lo;
    float cm = cmean[b * CH + c];
    float ci = cistd[b * CH + c];
    size_t base = ((size_t)b * CH + c) * HWSZ + q0 + hi * 8;
    #pragma unroll
    for (int r = 0; r < 8; ++r) {
      float mn = accM[t][r];
      float sd = sqrtf(fmaxf(accS[t][r] - mn * mn, 0.f));
      float cv = content[base + r];
      out[base + r] = sd * (cv - cm) * ci + mn;
    }
  }
}

// ---------------------------------------------------------------------------
extern "C" void kernel_launch(void* const* d_in, const int* in_sizes, int n_in,
                              void* d_out, int out_size, void* d_ws, size_t ws_size,
                              hipStream_t stream) {
  const float* content     = (const float*)d_in[0];
  const float* style       = (const float*)d_in[1];
  const float* content_key = (const float*)d_in[2];
  const float* style_key   = (const float*)d_in[3];
  const float* f_w = (const float*)d_in[4];
  const float* f_b = (const float*)d_in[5];
  const float* g_w = (const float*)d_in[6];
  const float* g_b = (const float*)d_in[7];
  const float* h_w = (const float*)d_in[8];
  const float* h_b = (const float*)d_in[9];
  float* out = (float*)d_out;

  // workspace carve-up (~69 MB)
  char* ws = (char*)d_ws;
  size_t off = 0;
  auto carve = [&](size_t bytes) -> char* {
    char* p = ws + off;
    off = (off + bytes + 255) & ~(size_t)255;
    return p;
  };
  const size_t big = (size_t)BATCH * HWSZ * CH * sizeof(unsigned short);  // 16 MB
  unsigned short* Qw  = (unsigned short*)carve(big);
  unsigned short* Ktw = (unsigned short*)carve(big);
  unsigned short* Vw  = (unsigned short*)carve(big);
  unsigned short* V2w = (unsigned short*)carve(big);
  unsigned short* wf  = (unsigned short*)carve((size_t)CH * CH * 2);
  unsigned short* wg  = (unsigned short*)carve((size_t)CH * CH * 2);
  unsigned short* wh  = (unsigned short*)carve((size_t)CH * CH * 2);
  float* cmeanw = (float*)carve((size_t)BATCH * CH * 4);
  float* cistdw = (float*)carve((size_t)BATCH * CH * 4);

  // 0) weights -> bf16
  wconv_kernel<<<(CH * CH + 255) / 256, 256, 0, stream>>>(f_w, g_w, h_w, wf, wg, wh);

  // 1) content statistics
  stats_kernel<<<BATCH * CH, 256, 0, stream>>>(content, cmeanw, cistdw);

  // 2) projections: Q = f(content_key), K^T = g(style_key) (query-major);
  //    V = h(style), V2 = V*V (channel-major)
  dim3 pgrid(HWSZ / 128, CH / 16, BATCH);
  proj_kernel<<<pgrid, 256, 0, stream>>>(content_key, wf, f_b, Qw,  nullptr, 0);
  proj_kernel<<<pgrid, 256, 0, stream>>>(style_key,   wg, g_b, Ktw, nullptr, 1);
  proj_kernel<<<pgrid, 256, 0, stream>>>(style,       wh, h_b, Vw,  V2w,     2);

  // 3) fused attention + AdaIN (needs ~148 KB dynamic LDS; fine on 320 KB/WGP)
  (void)hipFuncSetAttribute((const void*)attn_kernel,
                            hipFuncAttributeMaxDynamicSharedMemorySize, ATTN_LDS_BYTES);
  dim3 agrid(HWSZ / 16, BATCH);
  attn_kernel<<<agrid, 256, ATTN_LDS_BYTES, stream>>>(Qw, Ktw, Vw, V2w, content,
                                                      cmeanw, cistdw, out);
  (void)in_sizes; (void)n_in; (void)out_size; (void)ws_size;
}
